// TemporalDecoderHierarchical_3118146257159
// MI455X (gfx1250) — matrix operand order repaired and checked by reading, not compiled, not run on hardware
//
#include <hip/hip_runtime.h>
#include <hip/hip_bf16.h>
#include <math.h>

typedef __attribute__((ext_vector_type(16))) _Float16 v16h;
typedef __attribute__((ext_vector_type(8)))  _Float16 v8h;
typedef __attribute__((ext_vector_type(8)))  float    v8f;
typedef __attribute__((ext_vector_type(4)))  int      v4i;

#define WMMA_F16(a, b, c) \
    __builtin_amdgcn_wmma_f32_16x16x32_f16(false, (a), false, (b), (short)0, (c), false, false)

#if defined(__has_builtin)
#if __has_builtin(__builtin_amdgcn_global_load_async_to_lds_b128)
#define HAVE_ASYNC_LDS 1
#endif
#endif

// One 16-byte async global->LDS copy (gfx1250 GLOBAL_LOAD_ASYNC_TO_LDS_B128),
// falling back to load+ds_store if the builtin is unavailable.
__device__ __forceinline__ void stage16B(const _Float16* gsrc, _Float16* ldst) {
#ifdef HAVE_ASYNC_LDS
    __builtin_amdgcn_global_load_async_to_lds_b128(
        (__attribute__((address_space(1))) v4i*)gsrc,
        (__attribute__((address_space(3))) v4i*)ldst, 0, 0);
#else
    *(v8h*)ldst = *(const v8h*)gsrc;
#endif
}
__device__ __forceinline__ void stage_wait() {
#ifdef HAVE_ASYNC_LDS
    asm volatile("s_wait_asynccnt 0" ::: "memory");
#endif
}

// A-operand load (16-bit 16x32 layout): lane holds row M=lane&15;
// elements 0..15 = K at p[0..16).
// Caller passes p = rowptr + kchunk + (lane>>4)*16.
__device__ __forceinline__ v16h load_a16(const _Float16* p) {
    return *(const v16h*)p;
}

// Half-wave (16 lane) reductions: xor masks 1..8 stay inside lanes 0-15 / 16-31.
__device__ __forceinline__ float hmax16(float v) {
    v = fmaxf(v, __shfl_xor(v, 1, 32));
    v = fmaxf(v, __shfl_xor(v, 2, 32));
    v = fmaxf(v, __shfl_xor(v, 4, 32));
    v = fmaxf(v, __shfl_xor(v, 8, 32));
    return v;
}
__device__ __forceinline__ float hsum16(float v) {
    v += __shfl_xor(v, 1, 32);
    v += __shfl_xor(v, 2, 32);
    v += __shfl_xor(v, 4, 32);
    v += __shfl_xor(v, 8, 32);
    return v;
}

// ---------------------------------------------------------------- conversion
__global__ void cvt_f32_to_f16(const float* __restrict__ x, _Float16* __restrict__ y, int n) {
    int i = blockIdx.x * blockDim.x + threadIdx.x;
    if (i < n) y[i] = (_Float16)x[i];
}

// ---------------------------------------------------------------- GEMM C = A @ W^T
// A: [M,1024] f16 row-major.  W: [1024,1024] f16 row-major. C[m,n]=sum_k A[m,k]W[n,k].
// Block = 256 thr (8 waves) computes 256 rows x 64 cols (each wave: 32x64, 8 accums).
// The 64x32 W chunk is shared by all waves, staged through double-buffered LDS with
// async-to-LDS copies (1 b128/thread); k-loop manually unrolled x2 with fixed buffers.
// LDS rows padded to 40 halfs (80B) -> conflict-free 32B B-operand reads.
// mode 0: store f16 row-major [M,1024]
// mode 1: store f16 head-transposed Vt[b][h][d][s]  (M=B*S rows, S=4096)
// mode 2: store f32 row-major [M,1024]
__global__ void __launch_bounds__(256, 1)
gemm_xwT(const _Float16* __restrict__ A, const _Float16* __restrict__ W,
         _Float16* __restrict__ Ch, float* __restrict__ Cf, int M, int mode) {
    __shared__ __align__(16) _Float16 Wsh[2][64 * 40];

    const int tid  = threadIdx.x;
    const int lane = tid & 31;
    const int w    = tid >> 5;
    const int g    = lane >> 4;
    const int ln   = lane & 15;

    const int bm = (int)blockIdx.x >> 4;       // 256-row tile
    const int tn = (int)blockIdx.x & 15;       // 64-col tile
    const int row0 = bm * 256 + w * 32;

    // staging: thread tid covers W row (tn*64 + tid/4), 8 halfs at (tid%4)*8
    const _Float16* wsrc = W + (size_t)(tn * 64 + (tid >> 2)) * 1024 + (tid & 3) * 8;
    const int wdst = (tid >> 2) * 40 + (tid & 3) * 8;

    const _Float16* arow0 = A + (size_t)(row0 + ln) * 1024;
    const _Float16* arow1 = arow0 + (size_t)16 * 1024;

    v8f acc[2][4] = {};

    auto compute = [&](const _Float16* buf, int kc) {
        v16h a0 = load_a16(arow0 + kc * 32 + g * 16);
        v16h a1 = load_a16(arow1 + kc * 32 + g * 16);
        const _Float16* bsh = buf + ln * 40 + g * 16;
        v16h b0 = *(const v16h*)(bsh);
        v16h b1 = *(const v16h*)(bsh + 640);
        v16h b2 = *(const v16h*)(bsh + 1280);
        v16h b3 = *(const v16h*)(bsh + 1920);
        acc[0][0] = WMMA_F16(a0, b0, acc[0][0]);
        acc[0][1] = WMMA_F16(a0, b1, acc[0][1]);
        acc[0][2] = WMMA_F16(a0, b2, acc[0][2]);
        acc[0][3] = WMMA_F16(a0, b3, acc[0][3]);
        acc[1][0] = WMMA_F16(a1, b0, acc[1][0]);
        acc[1][1] = WMMA_F16(a1, b1, acc[1][1]);
        acc[1][2] = WMMA_F16(a1, b2, acc[1][2]);
        acc[1][3] = WMMA_F16(a1, b3, acc[1][3]);
    };

    stage16B(wsrc, &Wsh[0][wdst]);             // k-chunk 0 -> buf0
    stage_wait();
    __syncthreads();

#pragma unroll 1
    for (int kc = 0; kc < 32; kc += 2) {       // 1024 / 32 chunks, 2 per trip
        stage16B(wsrc + (kc + 1) * 32, &Wsh[1][wdst]);   // next chunk -> buf1
        compute(&Wsh[0][0], kc);
        stage_wait();
        __syncthreads();

        if (kc + 2 < 32)
            stage16B(wsrc + (kc + 2) * 32, &Wsh[0][wdst]);  // chunk after -> buf0
        compute(&Wsh[1][0], kc + 1);
        if (kc + 2 < 32) {
            stage_wait();
            __syncthreads();
        }
    }

#pragma unroll
    for (int p = 0; p < 2; p++)
#pragma unroll
        for (int t = 0; t < 4; t++)
#pragma unroll
            for (int r = 0; r < 8; r++) {
                int m = row0 + p * 16 + r + 8 * g;     // C/D layout: row = r + 8*(lane>=16)
                int n = tn * 64 + t * 16 + ln;         // col = lane&15 within tile
                float v = acc[p][t][r];
                if (mode == 0) {
                    Ch[(size_t)m * 1024 + n] = (_Float16)v;
                } else if (mode == 1) {
                    int bb = m >> 12, s = m & 4095;    // S = 4096
                    int hh = n >> 6,  d = n & 63;
                    Ch[(((size_t)bb * 16 + hh) * 64 + d) * 4096 + s] = (_Float16)v;
                } else {
                    Cf[(size_t)m * 1024 + n] = v;
                }
            }
}

// ---------------------------------------------------------------- fused flash attention
// One wave owns a 16-row Q tile of one (b,h). Loops S=4096 in chunks of 32.
// K tiles for chunk s+32 and V tiles for chunk s are prefetched right after the score
// WMMAs issue, so the softmax VALU work hides their latency.
__global__ void __launch_bounds__(256, 1)
attn_flash(const _Float16* __restrict__ Qp, const _Float16* __restrict__ Kp,
           const _Float16* __restrict__ Vt, const float* __restrict__ imp,
           _Float16* __restrict__ Y) {
    __shared__ __align__(16) _Float16 Psh[8][16 * 32];   // per-wave private P tile

    const int lane  = threadIdx.x & 31;
    const int wslot = threadIdx.x >> 5;
    const int wid   = blockIdx.x * 8 + wslot;
    const int qt    = wid & 127;            // 2048/16 q tiles
    const int h     = (wid >> 7) & 15;
    const int b     = wid >> 11;
    const int g     = lane >> 4;
    const int ln    = lane & 15;
    const int q0    = qt * 16;

    const _Float16* qrow = Qp + (size_t)(b * 2048 + q0 + ln) * 1024 + h * 64;
    v16h aq0 = load_a16(qrow + g * 16);
    v16h aq1 = load_a16(qrow + 32 + g * 16);

    // sensor index of every score column this lane touches == lane&15 (N_s==16, 16-aligned tiles)
    const float bias  = logf(fmaxf(imp[b * 16 + ln], 1e-6f));
    const float scale = 0.125f;             // 1/sqrt(64)

    float mrun[8], lrun[8];
#pragma unroll
    for (int r = 0; r < 8; r++) { mrun[r] = -3.0e38f; lrun[r] = 0.0f; }
    v8f yacc[4] = {};

    const _Float16* kbase = Kp + (size_t)(b * 4096) * 1024 + h * 64;
    const _Float16* vbase = Vt + ((size_t)(b * 16 + h) * 64) * 4096;

    auto loadK = [&](int s, v16h* bk) {
        const _Float16* k0 = kbase + (size_t)(s + ln) * 1024;
        const _Float16* k1 = k0 + (size_t)16 * 1024;
        bk[0] = *(const v16h*)(k0 + g * 16);
        bk[1] = *(const v16h*)(k1 + g * 16);
        bk[2] = *(const v16h*)(k0 + 32 + g * 16);
        bk[3] = *(const v16h*)(k1 + 32 + g * 16);
    };

    v16h bk[4];
    loadK(0, bk);

#pragma unroll 1
    for (int s = 0; s < 4096; s += 32) {
        // ---- scores: two 16x16 tiles covering columns [s, s+32)
        v8f s0v = {}, s1v = {};
        s0v = WMMA_F16(aq0, bk[0], s0v);
        s1v = WMMA_F16(aq0, bk[1], s1v);
        s0v = WMMA_F16(aq1, bk[2], s0v);
        s1v = WMMA_F16(aq1, bk[3], s1v);

        // ---- prefetch V (this chunk) and K (next chunk); softmax hides the latency
        v16h bv[4];
#pragma unroll
        for (int t = 0; t < 4; t++)
            bv[t] = *(const v16h*)(vbase + (size_t)(t * 16 + ln) * 4096 + s + g * 16);
        v16h nk[4];
        loadK((s + 32) & 4095, nk);

        // ---- online softmax update (rows live in 16-lane halves)
#pragma unroll
        for (int r = 0; r < 8; r++) {
            float v0 = s0v[r] * scale + bias;
            float v1 = s1v[r] * scale + bias;
            float cm = hmax16(fmaxf(v0, v1));
            float mn = fmaxf(mrun[r], cm);
            float alpha = __expf(mrun[r] - mn);
            float p0 = __expf(v0 - mn);
            float p1 = __expf(v1 - mn);
            lrun[r] = lrun[r] * alpha + hsum16(p0 + p1);
            mrun[r] = mn;
            yacc[0][r] *= alpha;
            yacc[1][r] *= alpha;
            yacc[2][r] *= alpha;
            yacc[3][r] *= alpha;
            int row = r + 8 * g;
            Psh[wslot][row * 32 + ln]      = (_Float16)p0;   // C-layout -> LDS
            Psh[wslot][row * 32 + 16 + ln] = (_Float16)p1;
        }
        // LDS is in-order per wave; wait + memory clobber stops compiler reordering.
        asm volatile("s_wait_dscnt 0" ::: "memory");

        // ---- reload P in A layout, multiply with V chunk (32 x 64)
        v16h ap = load_a16(&Psh[wslot][ln * 32 + g * 16]);
#pragma unroll
        for (int t = 0; t < 4; t++)
            yacc[t] = WMMA_F16(ap, bv[t], yacc[t]);
        asm volatile("s_wait_dscnt 0" ::: "memory");   // next iter's P stores follow these loads

        bk[0] = nk[0]; bk[1] = nk[1]; bk[2] = nk[2]; bk[3] = nk[3];
    }

    // ---- normalize and emit f16 y in merged-head row-major [B*T_q, 1024]
    _Float16* yout = Y + (size_t)(b * 2048 + q0) * 1024 + h * 64;
#pragma unroll
    for (int t = 0; t < 4; t++)
#pragma unroll
        for (int r = 0; r < 8; r++) {
            float val = yacc[t][r] / lrun[r];
            yout[(size_t)(r + 8 * g) * 1024 + t * 16 + ln] = (_Float16)val;
        }
}

// ---------------------------------------------------------------- host launcher
extern "C" void kernel_launch(void* const* d_in, const int* in_sizes, int n_in,
                              void* d_out, int out_size, void* d_ws, size_t ws_size,
                              hipStream_t stream) {
    const float* query = (const float*)d_in[0];   // [2,2048,1024]
    const float* kv    = (const float*)d_in[1];   // [2,256,16,1024] -> [2,4096,1024]
    const float* imp   = (const float*)d_in[2];   // [2,16]
    const float* Wq    = (const float*)d_in[3];
    const float* Wk    = (const float*)d_in[4];
    const float* Wv    = (const float*)d_in[5];
    const float* Wo    = (const float*)d_in[6];

    char* ws = (char*)d_ws;
    const size_t MB = 1u << 20;
    _Float16* q16  = (_Float16*)(ws + 0 * MB);    //  8 MB  [4096,1024]
    _Float16* kv16 = (_Float16*)(ws + 8 * MB);    // 16 MB  [8192,1024]
    _Float16* Wq16 = (_Float16*)(ws + 24 * MB);   //  2 MB
    _Float16* Wk16 = (_Float16*)(ws + 26 * MB);   //  2 MB
    _Float16* Wv16 = (_Float16*)(ws + 28 * MB);   //  2 MB
    _Float16* Wo16 = (_Float16*)(ws + 30 * MB);   //  2 MB
    _Float16* Qp   = (_Float16*)(ws + 32 * MB);   //  8 MB  [4096,1024]
    _Float16* Kp   = (_Float16*)(ws + 40 * MB);   // 16 MB  [8192,1024]
    _Float16* Vt   = (_Float16*)(ws + 56 * MB);   // 16 MB  [B,h,64,4096]
    _Float16* y16  = (_Float16*)(ws + 72 * MB);   //  8 MB  [4096,1024]

    const int nq = 2 * 2048 * 1024;
    const int nk = 2 * 4096 * 1024;
    const int nw = 1024 * 1024;
    cvt_f32_to_f16<<<(nq + 255) / 256, 256, 0, stream>>>(query, q16, nq);
    cvt_f32_to_f16<<<(nk + 255) / 256, 256, 0, stream>>>(kv, kv16, nk);
    cvt_f32_to_f16<<<(nw + 255) / 256, 256, 0, stream>>>(Wq, Wq16, nw);
    cvt_f32_to_f16<<<(nw + 255) / 256, 256, 0, stream>>>(Wk, Wk16, nw);
    cvt_f32_to_f16<<<(nw + 255) / 256, 256, 0, stream>>>(Wv, Wv16, nw);
    cvt_f32_to_f16<<<(nw + 255) / 256, 256, 0, stream>>>(Wo, Wo16, nw);

    // Projections: blocks = (M/256) * 16
    gemm_xwT<<<256, 256, 0, stream>>>(q16,  Wq16, Qp, nullptr, 4096, 0);
    gemm_xwT<<<512, 256, 0, stream>>>(kv16, Wk16, Kp, nullptr, 8192, 0);
    gemm_xwT<<<512, 256, 0, stream>>>(kv16, Wv16, Vt, nullptr, 8192, 1);

    // Fused attention: 2*16*128 = 4096 waves
    attn_flash<<<512, 256, 0, stream>>>(Qp, Kp, Vt, imp, y16);

    // Output projection -> fp32 d_out [2,2048,1024]
    gemm_xwT<<<256, 256, 0, stream>>>(y16, Wo16, nullptr, (float*)d_out, 4096, 2);
}
